// SelfBlockTSSA_23493471109513
// MI455X (gfx1250) — matrix operand order repaired and checked by reading, not compiled, not run on hardware
//
#include <hip/hip_runtime.h>
#include <cstdint>
#include <cstddef>

// ---------------------------------------------------------------------------
// Types for CDNA5 WMMA
// ---------------------------------------------------------------------------
typedef __bf16 bf16_t;
typedef __attribute__((ext_vector_type(16))) __bf16 v16bf;
typedef __attribute__((ext_vector_type(8)))  float  v8f;

// Problem constants
#define PB   16
#define PN   4096
#define PC   256
#define PH   4
#define PD   64
#define PROWS (PB * PN)        // 65536

// ---------------------------------------------------------------------------
// CDNA5 async global->LDS copy (tracked by ASYNCcnt), 16B per lane.
// ISA §15.18: GLOBAL_LOAD_ASYNC_TO_LDS_B128  vdst(LDS addr VGPR), vaddr, off
// LDS byte address = low 32 bits of the generic pointer (flat LDS aperture).
// ---------------------------------------------------------------------------
__device__ __forceinline__ void async_copy_b128(void* lds, const void* gptr)
{
    uint32_t lds_off = (uint32_t)(uintptr_t)lds;
    asm volatile("global_load_async_to_lds_b128 %0, %1, off"
                 :: "v"(lds_off), "v"(gptr) : "memory");
}
__device__ __forceinline__ void async_wait_all()
{
    asm volatile("s_wait_asynccnt 0x0" ::: "memory");
}
__device__ __forceinline__ void async_wait_3()
{
    asm volatile("s_wait_asynccnt 0x3" ::: "memory");
}

// ---------------------------------------------------------------------------
// Generic bf16 WMMA GEMM:  C[m,n] = sum_k A[m,k] * W[n,k] (+bias[n]) (+resid)
// A: bf16 [M, lda] row-major; W: bf16 [N, ldw] row-major (i.e. B = W^T).
// Block tile 256x128, 8 waves, each wave a 64x64 tile (4x4 WMMA tiles).
// Double-buffered LDS filled with async-to-LDS copies; last iter peeled so
// the steady-state K-loop is branch-free.
// Caller guarantees M%256==0, N%128==0, K%32==0 (and K>=64), 16B alignment.
// ---------------------------------------------------------------------------
#define TBM 256
#define TBN 128
#define TBK 32
#define LDSS 40   // padded LDS row stride (bf16 elems): 80B = 20 banks

__global__ __launch_bounds__(256)
void gemm_bf16_wmma(const bf16_t* __restrict__ A, int lda,
                    const bf16_t* __restrict__ W, int ldw,
                    float* __restrict__ Cf, bf16_t* __restrict__ Cb, int ldc,
                    const float* __restrict__ bias,
                    const float* __restrict__ resid,
                    int K)
{
    __shared__ bf16_t As[2][TBM * LDSS];
    __shared__ bf16_t Bs[2][TBN * LDSS];

    const int tid  = threadIdx.x;
    const int m0   = blockIdx.x * TBM;
    const int n0   = blockIdx.y * TBN;
    const int wid  = tid >> 5;        // 8 waves
    const int lane = tid & 31;
    const int wm   = wid >> 1;        // 0..3 : 64-row band
    const int wn   = wid & 1;         // 0..1 : 64-col band
    const int lh   = lane >> 4;       // 0/1
    const int lr   = lane & 15;

    v8f zero = {};
    v8f acc[4][4];
#pragma unroll
    for (int mi = 0; mi < 4; ++mi)
#pragma unroll
        for (int ni = 0; ni < 4; ++ni) acc[mi][ni] = zero;

    // A staging: thread t owns row t of the 256-row tile (2 x 16B = bf16 k 0..7 and 16..23)
    const bf16_t* gA = A + (size_t)(m0 + tid) * lda;
    bf16_t* lA[2] = { &As[0][tid * LDSS], &As[1][tid * LDSS] };
    // B staging: thread t copies 16B (row t/2, 8 bf16 from k (t&1)*16)
    const int  brow = tid >> 1, bcol = (tid & 1) * 16;
    const bf16_t* gB = W + (size_t)(n0 + brow) * ldw + bcol;
    bf16_t* lB[2] = { &Bs[0][brow * LDSS + bcol], &Bs[1][brow * LDSS + bcol] };

    // per-wave frag base rows (hoisted)
    const int arow = wm * 64 + lr;
    const int brow2 = wn * 64 + lr;

    auto compute_tile = [&](int buf) {
        union FragU { uint4 u[2]; v16bf v; };
        FragU af[4], bfr[4];
        const bf16_t* Ab = &As[buf][0];
        const bf16_t* Bb = &Bs[buf][0];
        // A 16x32 bf16 frag: lanes0-15 -> K {0..7,16..23}; lanes16-31 -> {8..15,24..31}
#pragma unroll
        for (int mi = 0; mi < 4; ++mi) {
            const bf16_t* ar = Ab + (arow + mi * 16) * LDSS;
            af[mi].u[0] = *(const uint4*)(ar + lh * 8);
            af[mi].u[1] = *(const uint4*)(ar + 16 + lh * 8);
        }
        // B 32x16 bf16 frag: lanes0-15 -> K {0..7,16..23} of col n; lanes16-31 -> {8..15,24..31}
#pragma unroll
        for (int ni = 0; ni < 4; ++ni) {
            const bf16_t* br = Bb + (brow2 + ni * 16) * LDSS;
            bfr[ni].u[0] = *(const uint4*)(br + lh * 8);
            bfr[ni].u[1] = *(const uint4*)(br + 16 + lh * 8);
        }
#pragma unroll
        for (int mi = 0; mi < 4; ++mi)
#pragma unroll
            for (int ni = 0; ni < 4; ++ni)
                acc[mi][ni] = __builtin_amdgcn_wmma_f32_16x16x32_bf16(
                    false, af[mi].v, false, bfr[ni].v,
                    (short)0, acc[mi][ni], false, false);
    };

    const int nsteps = K / TBK;

    // prologue: tile 0 -> buffer 0  (3 async instructions per wave)
    async_copy_b128(lA[0],      gA);
    async_copy_b128(lA[0] + 16, gA + 16);
    async_copy_b128(lB[0],      gB);

    // steady state: branch-free; always prefetch tile i+1 into the other buf
    for (int i = 0; i < nsteps - 1; ++i) {
        const int cur = i & 1;
        const int nxt = cur ^ 1;
        const int k1  = (i + 1) * TBK;
        async_copy_b128(lA[nxt],      gA + k1);
        async_copy_b128(lA[nxt] + 16, gA + k1 + 16);
        async_copy_b128(lB[nxt],      gB + k1);
        async_wait_3();           // tile i's 3 ops complete (FIFO per wave)
        __syncthreads();          // all waves' tile-i data visible
        compute_tile(cur);
        __syncthreads();          // reads of buf[cur] done before overwrite
    }

    // last tile
    async_wait_all();
    __syncthreads();
    compute_tile((nsteps - 1) & 1);

    // Epilogue. C layout: VGPR r, lanes0-15 -> M=r; lanes16-31 -> M=r+8; N=lane%16
#pragma unroll
    for (int mi = 0; mi < 4; ++mi) {
#pragma unroll
        for (int ni = 0; ni < 4; ++ni) {
            const int col = n0 + wn * 64 + ni * 16 + lr;
            const float bcol = bias ? bias[col] : 0.0f;
#pragma unroll
            for (int r = 0; r < 8; ++r) {
                const int row = m0 + wm * 64 + mi * 16 + lh * 8 + r;
                float v = acc[mi][ni][r] + bcol;
                if (resid) v += resid[(size_t)row * ldc + col];
                if (Cf) Cf[(size_t)row * ldc + col] = v;
                else    Cb[(size_t)row * ldc + col] = (bf16_t)v;
            }
        }
    }
}

// ---------------------------------------------------------------------------
// Conversion / small prep kernels
// ---------------------------------------------------------------------------
__global__ void cvt_f32_bf16(const float* __restrict__ src, bf16_t* __restrict__ dst, int n)
{
    int i = blockIdx.x * 256 + threadIdx.x;
    if (i < n) dst[i] = (bf16_t)src[i];
}

// x (B,N,C) f32 -> first half of cat buffer (B*N, 512) bf16
__global__ void cvt_x_to_cat(const float* __restrict__ x, bf16_t* __restrict__ cat)
{
    size_t i = (size_t)blockIdx.x * 256 + threadIdx.x;  // over B*N*C
    size_t bn = i >> 8;
    int c = (int)(i & 255);
    cat[bn * 512 + c] = (bf16_t)x[i];
}

// Reorder QKV weight rows so outputs are planar [q(256) | k(256) | v(256)].
__global__ void cvt_qkv_reorder(const float* __restrict__ Wsrc, const float* __restrict__ bsrc,
                                bf16_t* __restrict__ Wdst, float* __restrict__ bdst)
{
    int r = blockIdx.x;            // 0..767
    int t = threadIdx.x;           // 0..255
    int s = r >> 8;
    int hd = r & 255;
    int rin = hd * 3 + s;
    Wdst[(size_t)r * 256 + t] = (bf16_t)Wsrc[(size_t)rin * 256 + t];
    if (t == 0) bdst[r] = bsrc[rin];
}

// Wf = out_proj_w @ tssa_out_w  (256x256 @ 256x256), bf16 output
__global__ void fuse_w_kernel(const float* __restrict__ P, const float* __restrict__ T,
                              bf16_t* __restrict__ Wf)
{
    int i = blockIdx.x;            // row
    int j = threadIdx.x;           // col
    float acc = 0.0f;
#pragma unroll 8
    for (int k = 0; k < 256; ++k)
        acc += P[(size_t)i * 256 + k] * T[(size_t)k * 256 + j];
    Wf[(size_t)i * 256 + j] = (bf16_t)acc;
}

// biasF = out_proj_w @ tssa_out_b + out_proj_b
__global__ void fuse_b_kernel(const float* __restrict__ P, const float* __restrict__ tb,
                              const float* __restrict__ pb, float* __restrict__ bF)
{
    int i = threadIdx.x;           // 0..255
    float acc = pb[i];
#pragma unroll 8
    for (int k = 0; k < 256; ++k) acc += P[(size_t)i * 256 + k] * tb[k];
    bF[i] = acc;
}

__global__ void zero_f32(float* __restrict__ p, int n)
{
    int i = blockIdx.x * 256 + threadIdx.x;
    if (i < n) p[i] = 0.0f;
}

// ---------------------------------------------------------------------------
// RoPE + combine: w_input = (rope(q) + rope(k) + v) / 3, bf16 out
// qkv planar bf16 (B*N, 768); encoding f32 (2,B,1,N,D)
// ---------------------------------------------------------------------------
__global__ void rope_combine(const bf16_t* __restrict__ qkv, const float* __restrict__ enc,
                             bf16_t* __restrict__ wib)
{
    size_t p = (size_t)blockIdx.x * 256 + threadIdx.x;  // over B*N*C/2 pairs
    int cp = (int)(p & 127);
    size_t bn = p >> 7;
    int n = (int)(bn & (PN - 1));
    int b = (int)(bn >> 12);
    int c = cp * 2;
    int dd = c & (PD - 1);

    const bf16_t* base = qkv + bn * 768;
    float q0 = (float)base[c],       q1 = (float)base[c + 1];
    float k0 = (float)base[256 + c], k1 = (float)base[256 + c + 1];
    float v0 = (float)base[512 + c], v1 = (float)base[512 + c + 1];

    const float* e0 = enc + ((size_t)b * PN + n) * PD + dd;
    const float* e1 = e0 + (size_t)PB * PN * PD;
    float e0a = e0[0], e0b = e0[1], e1a = e1[0], e1b = e1[1];

    float qr0 = q0 * e0a - q1 * e1a, qr1 = q1 * e0b + q0 * e1b;
    float kr0 = k0 * e0a - k1 * e1a, kr1 = k1 * e0b + k0 * e1b;

    const float third = 1.0f / 3.0f;
    wib[bn * 256 + c]     = (bf16_t)((qr0 + kr0 + v0) * third);
    wib[bn * 256 + c + 1] = (bf16_t)((qr1 + kr1 + v1) * third);
}

// ---------------------------------------------------------------------------
// norm2[b, c] = sum_n w[b,n,c]^2   (L2 over sequence axis), chunked + atomics
// ---------------------------------------------------------------------------
__global__ __launch_bounds__(256)
void colnorm2_kernel(const float* __restrict__ w, float* __restrict__ norm2)
{
    int b = blockIdx.x, chunk = blockIdx.y, t = threadIdx.x;
    const float* base = w + ((size_t)b * PN + (size_t)chunk * 128) * PC + t;
    float acc = 0.0f;
    for (int i = 0; i < 128; ++i) {
        float v = base[(size_t)i * PC];
        acc += v * v;
    }
    atomicAdd(&norm2[b * PC + t], acc);
}

// ---------------------------------------------------------------------------
// Per-(b,n) TSSA stats: head softmax -> Pi; accumulate PiSum/DotRaw.
// One wave per (b,n); lane owns 8 consecutive channels; head = lane/8.
// ---------------------------------------------------------------------------
__global__ __launch_bounds__(256)
void tssa_stats_kernel(const float* __restrict__ w, const float* __restrict__ norm2,
                       const float* __restrict__ temp,
                       float* __restrict__ Pi, float* __restrict__ PiSum,
                       float* __restrict__ DotRaw)
{
    __shared__ float sPi[PH], sDot[PH];
    if (threadIdx.x < PH) { sPi[threadIdx.x] = 0.0f; sDot[threadIdx.x] = 0.0f; }
    __syncthreads();

    const int wid  = threadIdx.x >> 5;
    const int lane = threadIdx.x & 31;
    const size_t bn = (size_t)blockIdx.x * 8 + wid;
    const int b = (int)(bn >> 12);
    const int n = (int)(bn & (PN - 1));

    const float* wr = w + bn * PC + lane * 8;
    const float* nr = norm2 + b * PC + lane * 8;

    float s = 0.0f, q = 0.0f;
#pragma unroll
    for (int i = 0; i < 8; ++i) {
        float v = wr[i];
        float vv = v * v;
        q += vv;
        s += vv / fmaxf(nr[i], 1e-24f);   // == (w/max(||w||,1e-12))^2
    }
#pragma unroll
    for (int m = 1; m < 8; m <<= 1) {
        s += __shfl_xor(s, m, 32);
        q += __shfl_xor(q, m, 32);
    }
    float s0 = __shfl(s, 0, 32), s1 = __shfl(s, 8, 32);
    float s2 = __shfl(s, 16, 32), s3 = __shfl(s, 24, 32);
    float e0 = s0 * temp[0], e1 = s1 * temp[1], e2 = s2 * temp[2], e3 = s3 * temp[3];
    float mx = fmaxf(fmaxf(e0, e1), fmaxf(e2, e3));
    float p0 = __expf(e0 - mx), p1 = __expf(e1 - mx);
    float p2 = __expf(e2 - mx), p3 = __expf(e3 - mx);
    float den = p0 + p1 + p2 + p3;

    if ((lane & 7) == 0) {
        int h = lane >> 3;
        float ph = (h == 0 ? p0 : h == 1 ? p1 : h == 2 ? p2 : p3) / den;
        Pi[((size_t)(b * PH + h)) * PN + n] = ph;
        atomicAdd(&sPi[h], ph);
        atomicAdd(&sDot[h], ph * q);
    }
    __syncthreads();
    if (threadIdx.x < PH) {
        atomicAdd(&PiSum[b * PH + threadIdx.x], sPi[threadIdx.x]);
        atomicAdd(&DotRaw[b * PH + threadIdx.x], sDot[threadIdx.x]);
    }
}

// out[b,n,c] = -(w * Pi[b,h,n]) * attn[b,h] -> bf16
__global__ void tssa_out_kernel(const float* __restrict__ w, const float* __restrict__ Pi,
                                const float* __restrict__ PiSum, const float* __restrict__ DotRaw,
                                bf16_t* __restrict__ outb)
{
    size_t i = (size_t)blockIdx.x * 256 + threadIdx.x;  // over B*N*C
    int c = (int)(i & 255);
    size_t bn = i >> 8;
    int n = (int)(bn & (PN - 1));
    int b = (int)(bn >> 12);
    int h = c >> 6;
    float dots = DotRaw[b * PH + h] / (PiSum[b * PH + h] + 1e-8f);
    float attn = 1.0f / (1.0f + dots);
    float val = -w[i] * Pi[((size_t)(b * PH + h)) * PN + n] * attn;
    outb[i] = (bf16_t)val;
}

// ---------------------------------------------------------------------------
// LayerNorm (512) + exact GELU, in-place on bf16 h1. One block per row.
// ---------------------------------------------------------------------------
__global__ __launch_bounds__(128)
void ln_gelu_kernel(bf16_t* __restrict__ h1, const float* __restrict__ g,
                    const float* __restrict__ bta)
{
    const size_t row = blockIdx.x;
    bf16_t* hr = h1 + row * 512;
    const int t = threadIdx.x;
    const int lane = t & 31, wid = t >> 5;

    float v[4];
#pragma unroll
    for (int i = 0; i < 4; ++i) v[i] = (float)hr[t * 4 + i];

    float s1 = v[0] + v[1] + v[2] + v[3];
    float s2 = v[0] * v[0] + v[1] * v[1] + v[2] * v[2] + v[3] * v[3];
#pragma unroll
    for (int m = 1; m < 32; m <<= 1) {
        s1 += __shfl_xor(s1, m, 32);
        s2 += __shfl_xor(s2, m, 32);
    }
    __shared__ float a1[4], a2[4];
    if (lane == 0) { a1[wid] = s1; a2[wid] = s2; }
    __syncthreads();
    s1 = a1[0] + a1[1] + a1[2] + a1[3];
    s2 = a2[0] + a2[1] + a2[2] + a2[3];

    const float mu  = s1 * (1.0f / 512.0f);
    const float var = s2 * (1.0f / 512.0f) - mu * mu;
    const float rs  = rsqrtf(var + 1e-5f);
#pragma unroll
    for (int i = 0; i < 4; ++i) {
        int c = t * 4 + i;
        float x = (v[i] - mu) * rs * g[c] + bta[c];
        float ge = 0.5f * x * (1.0f + erff(x * 0.70710678118654752f));
        hr[c] = (bf16_t)ge;
    }
}

// ---------------------------------------------------------------------------
// Host side
// ---------------------------------------------------------------------------
extern "C" void kernel_launch(void* const* d_in, const int* in_sizes, int n_in,
                              void* d_out, int out_size, void* d_ws, size_t ws_size,
                              hipStream_t stream)
{
    (void)in_sizes; (void)n_in; (void)out_size; (void)ws_size;

    const float* x        = (const float*)d_in[0];
    const float* encoding = (const float*)d_in[1];
    const float* Wqkv_w   = (const float*)d_in[2];
    const float* Wqkv_b   = (const float*)d_in[3];
    const float* tqkv_w   = (const float*)d_in[4];
    const float* tqkv_b   = (const float*)d_in[5];
    const float* temp     = (const float*)d_in[6];
    const float* tout_w   = (const float*)d_in[7];
    const float* tout_b   = (const float*)d_in[8];
    const float* oproj_w  = (const float*)d_in[9];
    const float* oproj_b  = (const float*)d_in[10];
    const float* ffn1_w   = (const float*)d_in[11];
    const float* ffn1_b   = (const float*)d_in[12];
    const float* ln_g     = (const float*)d_in[13];
    const float* ln_b     = (const float*)d_in[14];
    const float* ffn2_w   = (const float*)d_in[15];
    const float* ffn2_b   = (const float*)d_in[16];
    float* out = (float*)d_out;

    // ---- workspace layout (256B-aligned bump allocator, deterministic) ----
    uint8_t* ws = (uint8_t*)d_ws;
    size_t off = 0;
    auto alloc = [&](size_t bytes) -> uint8_t* {
        uint8_t* p = ws + off;
        off = (off + bytes + 255) & ~(size_t)255;
        return p;
    };
    bf16_t* cat    = (bf16_t*)alloc((size_t)PROWS * 512 * 2);  // [x | message] bf16
    bf16_t* qkv    = (bf16_t*)alloc((size_t)PROWS * 768 * 2);  // planar q|k|v bf16
    bf16_t* wib    = (bf16_t*)alloc((size_t)PROWS * 256 * 2);  // w_input bf16
    float*  wf32   = (float*) alloc((size_t)PROWS * 256 * 4);  // w (f32)
    bf16_t* outb   = (bf16_t*)alloc((size_t)PROWS * 256 * 2);  // TSSA out bf16
    bf16_t* h1     = (bf16_t*)alloc((size_t)PROWS * 512 * 2);  // ffn1 out / gelu
    bf16_t* WqkvR  = (bf16_t*)alloc((size_t)768 * 256 * 2);
    float*  qkvB   = (float*) alloc(768 * 4);
    bf16_t* Wt     = (bf16_t*)alloc((size_t)256 * 256 * 2);    // tssa_qkv_w bf16
    bf16_t* Wf     = (bf16_t*)alloc((size_t)256 * 256 * 2);    // fused proj weight
    float*  bF     = (float*) alloc(256 * 4);                  // fused proj bias
    bf16_t* W1     = (bf16_t*)alloc((size_t)512 * 512 * 2);    // ffn1_w bf16
    bf16_t* W2     = (bf16_t*)alloc((size_t)256 * 512 * 2);    // ffn2_w bf16
    float*  stats  = (float*) alloc((4096 + 64 + 64) * 4);
    float*  norm2  = stats;            // [B,256]
    float*  PiSum  = stats + 4096;     // [B,H]
    float*  DotRaw = stats + 4096 + 64;
    float*  Pi     = (float*)alloc((size_t)PB * PH * PN * 4);

    // ---- prep ----
    cvt_x_to_cat<<<PROWS * PC / 256, 256, 0, stream>>>(x, cat);
    cvt_qkv_reorder<<<768, 256, 0, stream>>>(Wqkv_w, Wqkv_b, WqkvR, qkvB);
    cvt_f32_bf16<<<(256 * 256 + 255) / 256, 256, 0, stream>>>(tqkv_w, Wt, 256 * 256);
    cvt_f32_bf16<<<(512 * 512 + 255) / 256, 256, 0, stream>>>(ffn1_w, W1, 512 * 512);
    cvt_f32_bf16<<<(256 * 512 + 255) / 256, 256, 0, stream>>>(ffn2_w, W2, 256 * 512);
    fuse_w_kernel<<<256, 256, 0, stream>>>(oproj_w, tout_w, Wf);
    fuse_b_kernel<<<1, 256, 0, stream>>>(oproj_w, tout_b, oproj_b, bF);
    zero_f32<<<(4224 + 255) / 256, 256, 0, stream>>>(stats, 4224);

    // ---- GEMM1: qkv = x @ WqkvR^T  (65536 x 768 x 256) -> planar bf16 ----
    gemm_bf16_wmma<<<dim3(PROWS / TBM, 768 / TBN), 256, 0, stream>>>(
        cat, 512, WqkvR, 256, nullptr, qkv, 768, qkvB, nullptr, 256);

    // ---- RoPE + combine -> w_input bf16 ----
    rope_combine<<<PROWS * PC / 2 / 256, 256, 0, stream>>>(qkv, encoding, wib);

    // ---- GEMM2: w = w_input @ tssa_qkv_w^T  (65536 x 256 x 256) -> f32 ----
    gemm_bf16_wmma<<<dim3(PROWS / TBM, 256 / TBN), 256, 0, stream>>>(
        wib, 256, Wt, 256, wf32, nullptr, 256, tqkv_b, nullptr, 256);

    // ---- TSSA reductions ----
    colnorm2_kernel<<<dim3(PB, 32), 256, 0, stream>>>(wf32, norm2);
    tssa_stats_kernel<<<PROWS / 8, 256, 0, stream>>>(wf32, norm2, temp, Pi, PiSum, DotRaw);
    tssa_out_kernel<<<PROWS * PC / 256, 256, 0, stream>>>(wf32, Pi, PiSum, DotRaw, outb);

    // ---- GEMM3 (fused tssa_out+out_proj): message -> cat[:,256:512] bf16 ----
    gemm_bf16_wmma<<<dim3(PROWS / TBM, 256 / TBN), 256, 0, stream>>>(
        outb, 256, Wf, 256, nullptr, cat + 256, 512, bF, nullptr, 256);

    // ---- GEMM4: h1 = cat @ ffn1_w^T  (65536 x 512 x 512) -> bf16 ----
    gemm_bf16_wmma<<<dim3(PROWS / TBM, 512 / TBN), 256, 0, stream>>>(
        cat, 512, W1, 512, nullptr, h1, 512, ffn1_b, nullptr, 512);

    // ---- LayerNorm + GELU (in place on h1) ----
    ln_gelu_kernel<<<PROWS, 128, 0, stream>>>(h1, ln_g, ln_b);

    // ---- GEMM5: out = x + g @ ffn2_w^T + b  (65536 x 256 x 512) -> f32 ----
    gemm_bf16_wmma<<<dim3(PROWS / TBM, 256 / TBN), 256, 0, stream>>>(
        h1, 512, W2, 512, out, nullptr, 256, ffn2_b, x, 512);
}
